// RephineLayer_18107582120177
// MI455X (gfx1250) — compile-verified, matrix operand order checked by reading, not run on hardware
//
#include <hip/hip_runtime.h>
#include <math.h>

#define BGR   128      // graphs
#define NPV   96       // nodes per graph
#define NV    12288    // BGR*NPV
#define EPG   384      // edges per graph
#define NE    49152    // BGR*EPG
#define NFEAT 512
#define HID   128
#define NFILT 8
#define OUT   64

typedef __attribute__((ext_vector_type(16))) __bf16    v16bf;
typedef __attribute__((ext_vector_type(8)))  float     v8f;
typedef __attribute__((ext_vector_type(8)))  unsigned  v8u;

// pack two f32 into one dword holding two bf16 (truncation rounding)
__device__ __forceinline__ unsigned pk2(float lo, float hi) {
  unsigned ul = __builtin_bit_cast(unsigned, lo);
  unsigned uh = __builtin_bit_cast(unsigned, hi);
  return (uh & 0xffff0000u) | (ul >> 16);
}

// ---------------------------------------------------------------------------
// Kernel 0: convert Wf1 [512,128] f32 -> Wt [128,512] bf16 (fragment-major,
// K contiguous per output column). 128 KB, lives in L2 for the whole GEMM.
// ---------------------------------------------------------------------------
__global__ __launch_bounds__(256) void k_cvtW(const float* __restrict__ Wf1,
                                              unsigned short* __restrict__ Wt) {
  int i = blockIdx.x * 256 + threadIdx.x;      // over HID*NFEAT
  if (i >= HID * NFEAT) return;
  int n = i >> 9;            // / NFEAT
  int k = i & (NFEAT - 1);
  unsigned u = __builtin_bit_cast(unsigned, Wf1[(size_t)k * HID + n]);
  Wt[i] = (unsigned short)(u >> 16);
}

// ---------------------------------------------------------------------------
// Kernel 1: H = relu(X @ Wf1 + bf1)   [NV,512] x [512,128] -> [NV,128]
// Block = 4 waves = 32 rows x 128 cols. Wave tile = 16 rows x 64 cols
// (4 accumulators). Per K-step: A = 4 x global_load_b128 (f32, packed to bf16
// in registers), B = 8 x global_load_b128 (pre-converted bf16, L2-resident),
// 4 x v_wmma_f32_16x16x32_bf16. No LDS, no barriers in the hot loop.
// ---------------------------------------------------------------------------
__global__ __launch_bounds__(128) void k_gemm1(const float* __restrict__ X,
                                               const unsigned short* __restrict__ Wt,
                                               const float* __restrict__ bf1,
                                               float* __restrict__ H) {
  const int tid  = threadIdx.x;
  const int wave = tid >> 5;
  const int lane = tid & 31;
  const int rowBase = blockIdx.x * 32 + (wave >> 1) * 16;
  const int colBase = (wave & 1) * 64;
  const int m     = rowBase + (lane & 15);   // A-frag row for this lane
  const int khalf = (lane >> 4) * 8;         // K sub-offset per A/B layout
  const int nl    = lane & 15;

  v8f c[4];
#pragma unroll
  for (int t = 0; t < 4; ++t) c[t] = (v8f){0.f,0.f,0.f,0.f,0.f,0.f,0.f,0.f};

  for (int K0 = 0; K0 < NFEAT; K0 += 32) {
    // A fragment: lane holds row m, K = K0+khalf+{0..7} and +16+{0..7}
    const float* ap = X + (size_t)m * NFEAT + K0 + khalf;
    float4 a0 = *(const float4*)(ap);
    float4 a1 = *(const float4*)(ap + 4);
    float4 a2 = *(const float4*)(ap + 16);
    float4 a3 = *(const float4*)(ap + 20);
    v8u au = { pk2(a0.x, a0.y), pk2(a0.z, a0.w), pk2(a1.x, a1.y), pk2(a1.z, a1.w),
               pk2(a2.x, a2.y), pk2(a2.z, a2.w), pk2(a3.x, a3.y), pk2(a3.z, a3.w) };
    v16bf a = __builtin_bit_cast(v16bf, au);

#pragma unroll
    for (int t = 0; t < 4; ++t) {
      int n = colBase + t * 16 + nl;
      const unsigned short* bp = Wt + (size_t)n * NFEAT + K0 + khalf;
      uint4 blo = *(const uint4*)(bp);        // K khalf..khalf+7 (8 bf16)
      uint4 bhi = *(const uint4*)(bp + 16);   // K khalf+16..+23
      v8u bu = { blo.x, blo.y, blo.z, blo.w, bhi.x, bhi.y, bhi.z, bhi.w };
      v16bf b = __builtin_bit_cast(v16bf, bu);
      c[t] = __builtin_amdgcn_wmma_f32_16x16x32_bf16(false, a, false, b,
                                                     (short)0, c[t], false, false);
    }
  }

  // epilogue: C/D layout -> lane covers N = lane&15, M = r + 8*(lane>>4)
  const int mrow = 8 * (lane >> 4);
#pragma unroll
  for (int t = 0; t < 4; ++t) {
    int n = colBase + t * 16 + nl;
    float bias = bf1[n];
#pragma unroll
    for (int r = 0; r < 8; ++r) {
      int mm = rowBase + mrow + r;
      H[(size_t)mm * HID + n] = fmaxf(c[t][r] + bias, 0.f);
    }
  }
}

// ---------------------------------------------------------------------------
// Kernel 2: fv = sigmoid(H @ Wf2 + bf2)   [NV,128] x [128,8] -> [NV,8]
// ---------------------------------------------------------------------------
__global__ __launch_bounds__(256) void k_fv2(const float* __restrict__ H,
                                             const float* __restrict__ Wf2,
                                             const float* __restrict__ bf2,
                                             float* __restrict__ fv) {
  __shared__ float sw[HID * NFILT];
  const int tid = threadIdx.x;
  for (int i = tid; i < HID * NFILT; i += 256) sw[i] = Wf2[i];
  __syncthreads();
  const int v = blockIdx.x * 256 + tid;
  if (v >= NV) return;
  float acc[NFILT];
#pragma unroll
  for (int f = 0; f < NFILT; ++f) acc[f] = 0.f;
  const float* hr = H + (size_t)v * HID;
  for (int k = 0; k < HID; ++k) {
    float h = hr[k];
#pragma unroll
    for (int f = 0; f < NFILT; ++f) acc[f] += h * sw[k * NFILT + f];
  }
#pragma unroll
  for (int f = 0; f < NFILT; ++f) {
    float s = acc[f] + bf2[f];
    fv[(size_t)v * NFILT + f] = 1.0f / (1.0f + __expf(-s));
  }
}

// ---------------------------------------------------------------------------
// Kernel 3: fe = max(fv[u], fv[v]) per edge; gmax = per-graph max over edges
// One block per graph.
// ---------------------------------------------------------------------------
__global__ __launch_bounds__(128) void k_fe(const float* __restrict__ fv,
                                            const int* __restrict__ ei,
                                            float* __restrict__ fe,
                                            float* __restrict__ gmax) {
  __shared__ float red[128][NFILT];
  const int b = blockIdx.x, tid = threadIdx.x;
  float rm[NFILT];
#pragma unroll
  for (int f = 0; f < NFILT; ++f) rm[f] = -1e30f;
  for (int el = tid; el < EPG; el += 128) {
    int e = b * EPG + el;
    int u = ei[e], v = ei[NE + e];
#pragma unroll
    for (int f = 0; f < NFILT; ++f) {
      float x = fmaxf(fv[(size_t)u * NFILT + f], fv[(size_t)v * NFILT + f]);
      fe[(size_t)e * NFILT + f] = x;
      rm[f] = fmaxf(rm[f], x);
    }
  }
#pragma unroll
  for (int f = 0; f < NFILT; ++f) red[tid][f] = rm[f];
  __syncthreads();
  for (int s = 64; s > 0; s >>= 1) {
    if (tid < s)
#pragma unroll
      for (int f = 0; f < NFILT; ++f)
        red[tid][f] = fmaxf(red[tid][f], red[tid + s][f]);
    __syncthreads();
  }
  if (tid < NFILT) gmax[b * NFILT + tid] = red[0][tid];
}

// ---------------------------------------------------------------------------
// Kernel 4: persistence pairing. 1 wave per (graph, channel) task, 4 waves /
// block. Lanes rank-sort edges in LDS (stable, tie-break by index == argsort),
// lane 0 runs Kruskal + elder-rule union-find on LDS parent array.
// ---------------------------------------------------------------------------
__global__ __launch_bounds__(128) void k_pair(const float* __restrict__ fe,
                                              const float* __restrict__ fv,
                                              const int* __restrict__ ei,
                                              int* __restrict__ de,
                                              int* __restrict__ cyc) {
  __shared__ float sfe[4][EPG];
  __shared__ int   sord[4][EPG];
  __shared__ short su[4][EPG];
  __shared__ short sv[4][EPG];
  __shared__ float sfv[4][NPV];
  __shared__ int   spar[4][NPV];
  __shared__ int   sde[4][NPV];

  const int w = threadIdx.x >> 5, lane = threadIdx.x & 31;
  const int task = blockIdx.x * 4 + w;            // 0..1023
  const int g = task >> 3, f = task & 7;
  const int ebase = g * EPG, vbase = g * NPV;

  for (int i = lane; i < EPG; i += 32) {
    sfe[w][i] = fe[(size_t)(ebase + i) * NFILT + f];
    su[w][i]  = (short)(ei[ebase + i] - vbase);
    sv[w][i]  = (short)(ei[NE + ebase + i] - vbase);
  }
  for (int j = lane; j < NPV; j += 32) {
    sfv[w][j]  = fv[(size_t)(vbase + j) * NFILT + f];
    spar[w][j] = j;
    sde[w][j]  = -1;
  }
  __syncthreads();

  // stable rank sort (O(E^2)/32 per lane, all in LDS)
  for (int i = lane; i < EPG; i += 32) {
    float key = sfe[w][i];
    int r = 0;
    for (int j = 0; j < EPG; ++j) {
      float fj = sfe[w][j];
      r += (fj < key) || (fj == key && j < i);
    }
    sord[w][r] = i;
  }
  __syncthreads();

  // serial Kruskal with elder rule on lane 0 of each wave
  if (lane == 0) {
    for (int k = 0; k < EPG; ++k) {
      int e = sord[w][k];
      int u = su[w][e], v = sv[w][e];
      int ru = u; while (spar[w][ru] != ru) ru = spar[w][ru];
      int rv = v; while (spar[w][rv] != rv) rv = spar[w][rv];
      int isc = (ru == rv) ? 1 : 0;
      cyc[(size_t)(ebase + e) * NFILT + f] = isc;
      if (!isc) {
        int older   = (sfv[w][ru] <= sfv[w][rv]) ? ru : rv;
        int younger = ru + rv - older;
        sde[w][younger]  = ebase + e;   // global edge index that killed it
        spar[w][younger] = older;
      }
    }
  }
  __syncthreads();

  for (int j = lane; j < NPV; j += 32)
    de[(size_t)(vbase + j) * NFILT + f] = sde[w][j];
}

// ---------------------------------------------------------------------------
// Kernel 5: per-graph block: dim-1 DeepSet (x1), x0 assembly, DeepSetLayer0,
// z = g0 + x1
// ---------------------------------------------------------------------------
__global__ __launch_bounds__(256) void k_graph(const float* __restrict__ fv,
                                               const float* __restrict__ fe,
                                               const float* __restrict__ gmax,
                                               const int* __restrict__ de,
                                               const int* __restrict__ cyc,
                                               const float* __restrict__ Wg0,
                                               const float* __restrict__ bg0,
                                               const float* __restrict__ Wl0,
                                               const float* __restrict__ Wd1,
                                               const float* __restrict__ bd1,
                                               float* __restrict__ z) {
  __shared__ float x0s[NPV][2 * NFILT];
  __shared__ float xm[2 * NFILT];
  __shared__ float lam[OUT];
  __shared__ float g0s[OUT];
  __shared__ float x1s[OUT];
  __shared__ float sm0[NFILT];
  __shared__ float cnt[NFILT];

  const int b = blockIdx.x, tid = threadIdx.x;
  if (tid < NFILT) { sm0[tid] = 0.f; cnt[tid] = 0.f; }
  if (tid < OUT) g0s[tid] = 0.f;
  __syncthreads();

  // dim-1 stats: sum of fe over cycle edges + counts, per channel
  for (int t = tid; t < NFILT * EPG; t += 256) {
    int f = t / EPG, el = t - f * EPG;
    int e = b * EPG + el;
    if (cyc[(size_t)e * NFILT + f]) {
      atomicAdd(&sm0[f], fe[(size_t)e * NFILT + f]);
      atomicAdd(&cnt[f], 1.0f);
    }
  }
  // x0 assembly: (birth, death) interleaved per channel
  for (int t = tid; t < NPV * NFILT; t += 256) {
    int vl = t >> 3, f = t & 7;
    int vg = b * NPV + vl;
    int d = de[(size_t)vg * NFILT + f];
    float death = (d >= 0) ? fe[(size_t)d * NFILT + f] : gmax[b * NFILT + f];
    x0s[vl][2 * f]     = fv[(size_t)vg * NFILT + f];
    x0s[vl][2 * f + 1] = death;
  }
  __syncthreads();

  // x1 = (1/NFILT) * sum_f (mean_pers1_f @ Wd1 + bd1)
  if (tid < OUT) {
    float acc = 0.f;
    for (int f = 0; f < NFILT; ++f) {
      float c  = cnt[f];
      float dn = fmaxf(c, 1.0f);
      float m0 = sm0[f] / dn;
      float m1 = gmax[b * NFILT + f] * c / dn;   // sum of gmax over cyc edges / dn
      acc += m0 * Wd1[tid] + m1 * Wd1[OUT + tid] + bd1[tid];
    }
    x1s[tid] = acc * (1.0f / NFILT);
  }
  // column means of x0
  if (tid < 2 * NFILT) {
    float s = 0.f;
    for (int v = 0; v < NPV; ++v) s += x0s[v][tid];
    xm[tid] = s / (float)NPV;
  }
  __syncthreads();
  if (tid < OUT) {
    float s = 0.f;
    for (int c = 0; c < 2 * NFILT; ++c) s += xm[c] * Wl0[c * OUT + tid];
    lam[tid] = s;
  }
  __syncthreads();

  // h = relu(x0 @ Wg0 + bg0 - lam); g0 = mean-pool
  for (int t = tid; t < NPV * OUT; t += 256) {
    int v = t >> 6, o = t & 63;
    float s = bg0[o] - lam[o];
    for (int c = 0; c < 2 * NFILT; ++c) s += x0s[v][c] * Wg0[c * OUT + o];
    s = fmaxf(s, 0.f);
    atomicAdd(&g0s[o], s * (1.0f / NPV));
  }
  __syncthreads();
  if (tid < OUT) z[b * OUT + tid] = g0s[tid] + x1s[tid];
}

// ---------------------------------------------------------------------------
// Kernel 6: batch-norm (training stats) + 2-layer MLP.  Single workgroup.
// ---------------------------------------------------------------------------
__global__ __launch_bounds__(256) void k_final(const float* __restrict__ z,
                                               const float* __restrict__ Wo1,
                                               const float* __restrict__ bo1,
                                               const float* __restrict__ Wo2,
                                               const float* __restrict__ bo2,
                                               const float* __restrict__ bng,
                                               const float* __restrict__ bnb,
                                               float* __restrict__ out) {
  __shared__ float zn[BGR][OUT];
  __shared__ float y1[BGR][OUT];
  __shared__ float mu[OUT], iv[OUT];
  const int tid = threadIdx.x;
  for (int t = tid; t < BGR * OUT; t += 256) zn[t >> 6][t & 63] = z[t];
  __syncthreads();
  if (tid < OUT) {
    float s = 0.f;
    for (int r = 0; r < BGR; ++r) s += zn[r][tid];
    float m = s / (float)BGR;
    float v = 0.f;
    for (int r = 0; r < BGR; ++r) { float d = zn[r][tid] - m; v += d * d; }
    mu[tid] = m;
    iv[tid] = rsqrtf(v / (float)BGR + 1e-5f);
  }
  __syncthreads();
  for (int t = tid; t < BGR * OUT; t += 256) {
    int r = t >> 6, c = t & 63;
    zn[r][c] = (zn[r][c] - mu[c]) * iv[c] * bng[c] + bnb[c];
  }
  __syncthreads();
  for (int t = tid; t < BGR * OUT; t += 256) {
    int r = t >> 6, o = t & 63;
    float s = bo1[o];
    for (int k = 0; k < OUT; ++k) s += zn[r][k] * Wo1[k * OUT + o];
    y1[r][o] = fmaxf(s, 0.f);
  }
  __syncthreads();
  for (int t = tid; t < BGR * OUT; t += 256) {
    int r = t >> 6, o = t & 63;
    float s = bo2[o];
    for (int k = 0; k < OUT; ++k) s += y1[r][k] * Wo2[k * OUT + o];
    out[t] = s;
  }
}

// ---------------------------------------------------------------------------
extern "C" void kernel_launch(void* const* d_in, const int* in_sizes, int n_in,
                              void* d_out, int out_size, void* d_ws, size_t ws_size,
                              hipStream_t stream) {
  const float* x   = (const float*)d_in[0];
  const int*   ei  = (const int*)d_in[1];
  // d_in[2] vertex_slices, d_in[3] edge_slices, d_in[4] batch: implied by fixed layout
  const float* Wf1 = (const float*)d_in[5];
  const float* bf1 = (const float*)d_in[6];
  const float* Wf2 = (const float*)d_in[7];
  const float* bf2 = (const float*)d_in[8];
  const float* Wg0 = (const float*)d_in[9];
  const float* bg0 = (const float*)d_in[10];
  const float* Wl0 = (const float*)d_in[11];
  const float* Wd1 = (const float*)d_in[12];
  const float* bd1 = (const float*)d_in[13];
  const float* Wo1 = (const float*)d_in[14];
  const float* bo1 = (const float*)d_in[15];
  const float* Wo2 = (const float*)d_in[16];
  const float* bo2 = (const float*)d_in[17];
  const float* bng = (const float*)d_in[18];
  const float* bnb = (const float*)d_in[19];

  // workspace layout (~10.4 MB)
  float* H    = (float*)d_ws;                       // NV*HID
  float* fv   = H + (size_t)NV * HID;               // NV*NFILT
  float* fe   = fv + (size_t)NV * NFILT;            // NE*NFILT
  float* gmax = fe + (size_t)NE * NFILT;            // BGR*NFILT
  int*   de   = (int*)(gmax + BGR * NFILT);         // NV*NFILT
  int*   cyc  = de + (size_t)NV * NFILT;            // NE*NFILT
  float* z    = (float*)(cyc + (size_t)NE * NFILT); // BGR*OUT
  unsigned short* Wt = (unsigned short*)(z + BGR * OUT); // HID*NFEAT bf16

  k_cvtW<<<(HID * NFEAT + 255) / 256, 256, 0, stream>>>(Wf1, Wt);
  k_gemm1<<<NV / 32, 128, 0, stream>>>(x, Wt, bf1, H);
  k_fv2<<<(NV + 255) / 256, 256, 0, stream>>>(H, Wf2, bf2, fv);
  k_fe<<<BGR, 128, 0, stream>>>(fv, ei, fe, gmax);
  k_pair<<<(BGR * NFILT) / 4, 128, 0, stream>>>(fe, fv, ei, de, cyc);
  k_graph<<<BGR, 256, 0, stream>>>(fv, fe, gmax, de, cyc, Wg0, bg0, Wl0, Wd1, bd1, z);
  k_final<<<1, 256, 0, stream>>>(z, Wo1, bo1, Wo2, bo2, bng, bnb, (float*)d_out);
}
